// Encoder_45681272160862
// MI455X (gfx1250) — compile-verified, matrix operand order checked by reading, not run on hardware
//
#include <hip/hip_runtime.h>
#include <stdint.h>

// ---------------------------------------------------------------------------
// Types
// ---------------------------------------------------------------------------
typedef _Float16 half_t;
typedef __attribute__((ext_vector_type(16))) _Float16 v16h;
typedef __attribute__((ext_vector_type(8)))  _Float16 v8h;
typedef __attribute__((ext_vector_type(8)))  float    v8f;
typedef __attribute__((ext_vector_type(4)))  unsigned int u32x4;
typedef __attribute__((ext_vector_type(8)))  int      i32x8;
typedef __attribute__((ext_vector_type(4)))  int      i32x4;

#define T_STEPS   15
#define WIMG      1020
#define P1PAD     515     // padded pooled1 (511 + 2*2)
#define C2OUT     501
#define P2PAD     255     // padded pooled2 (251 + 2*2)
#define C3OUT     249
#define HW3       (249*249)          // 62001
#define N3        (15*2*249*249)     // one output tensor (spk or pot)

// ---------------------------------------------------------------------------
// WMMA wrapper (probe-confirmed 8-arg signature)
// ---------------------------------------------------------------------------
__device__ __forceinline__ v8f wmma16x16x32(v16h a, v16h b, v8f c) {
  return __builtin_amdgcn_wmma_f32_16x16x32_f16(false, a, false, b,
                                                (short)0, c, false, false);
}

// Unaligned 8xf16 LDS load (lane-shifted activation runs, 2B aligned).
__device__ __forceinline__ v8h load8u(const half_t* p) {
  v8h v; __builtin_memcpy(&v, p, sizeof(v)); return v;
}

#define CAT16(lo, hi) __builtin_shufflevector(lo, hi, 0,1,2,3,4,5,6,7,8,9,10,11,12,13,14,15)

// ---------------------------------------------------------------------------
// Tensor Data Mover: 3D tile (x fastest, then y, then channel) -> LDS.
// Builds D# groups per CDNA5 ISA. data_size = 2 bytes.
// ---------------------------------------------------------------------------
#if defined(__has_builtin)
#if __has_builtin(__builtin_amdgcn_tensor_load_to_lds) && \
    __has_builtin(__builtin_amdgcn_s_wait_tensorcnt)
#define USE_TDM 1
#endif
#endif

#ifdef USE_TDM
__device__ __forceinline__ void tdm_load_tile_f16(
    uint32_t lds_addr, const half_t* gptr,
    uint32_t tile_x, uint32_t tile_y, uint32_t tile_c,
    uint32_t tens_w, uint32_t tens_h, uint32_t tens_c,
    uint64_t row_stride, uint64_t ch_stride)
{
  uint64_t ga = (uint64_t)(uintptr_t)gptr;
  u32x4 g0;
  g0[0] = 1u;                                          // count=1 valid user D#
  g0[1] = lds_addr;                                    // lds_addr [63:32]
  g0[2] = (uint32_t)ga;                                // global_addr lo
  g0[3] = (uint32_t)((ga >> 32) & 0x1FFFFFFu) | (2u << 30);  // hi + type=2
  i32x8 g1;
  g1[0] = (int)(1u << 16);                             // data_size=1 (2B), mask=0
  g1[1] = (int)((tens_w & 0xFFFFu) << 16);             // tensor_dim0 lo16
  g1[2] = (int)((tens_w >> 16) | ((tens_h & 0xFFFFu) << 16));
  g1[3] = (int)((tens_h >> 16) | (tile_x << 16));      // tile_dim0
  g1[4] = (int)(tile_y | (tile_c << 16));              // tile_dim1, tile_dim2
  g1[5] = (int)(uint32_t)row_stride;                   // dim0 stride [31:0]
  g1[6] = (int)((uint32_t)((row_stride >> 32) & 0xFFFFu) |
                ((uint32_t)(ch_stride & 0xFFFFu) << 16));
  g1[7] = (int)(uint32_t)(ch_stride >> 16);            // dim1 stride hi
  i32x4 g2; g2[0] = (int)tens_c; g2[1] = 0; g2[2] = 0; g2[3] = 0;
  i32x4 g3 = {0, 0, 0, 0};
#if __clang_major__ >= 23
  i32x8 gx = {0, 0, 0, 0, 0, 0, 0, 0};
  __builtin_amdgcn_tensor_load_to_lds(g0, g1, g2, g3, gx, 0);
#else
  __builtin_amdgcn_tensor_load_to_lds(g0, g1, g2, g3, 0);
#endif
}
#endif

// ---------------------------------------------------------------------------
// Kernel 1: conv1 (pad2, 5x5, 2ch) + fire@5 + maxpool(2,2,pad1), fused.
// Writes pooled spikes as f16 into padded 515x515 buffer (border = 0).
// grid: (33, 33, 60=t*4+oc), block: (16,16)
// ---------------------------------------------------------------------------
__global__ __launch_bounds__(256) void conv1_fire_pool(
    const float* __restrict__ in, const float* __restrict__ w1,
    half_t* __restrict__ pooled1)
{
  __shared__ float sIn[2 * 36 * 36];
  __shared__ float sW[50];
  const int t  = blockIdx.z >> 2;
  const int oc = blockIdx.z & 3;
  const int by = blockIdx.y, bx = blockIdx.x;
  const int tid = threadIdx.y * 16 + threadIdx.x;

  if (tid < 50) sW[tid] = w1[oc * 50 + tid];

  const int iy0 = 32 * by - 7;
  const int ix0 = 32 * bx - 7;
  for (int idx = tid; idx < 2 * 36 * 36; idx += 256) {
    int ci = idx / 1296, rem = idx - ci * 1296;
    int r = rem / 36, cc = rem - r * 36;
    int iy = iy0 + r, ix = ix0 + cc;
    float v = 0.f;
    if ((unsigned)iy < (unsigned)WIMG && (unsigned)ix < (unsigned)WIMG)
      v = in[((t * 2 + ci) * WIMG + iy) * WIMG + ix];
    sIn[idx] = v;
  }
  __syncthreads();

  const int pyp = 16 * by + threadIdx.y;   // padded pooled coords
  const int pxp = 16 * bx + threadIdx.x;
  if (pyp >= P1PAD || pxp >= P1PAD) return;

  float spike = 0.f;
  if (pyp >= 2 && pyp < P1PAD - 2 && pxp >= 2 && pxp < P1PAD - 2) {
    const int py = pyp - 2, px = pxp - 2;  // pooled coords in [0,511)
    #pragma unroll
    for (int dy = 0; dy < 2; ++dy) {
      int yc = 2 * py - 1 + dy;            // conv1 output row
      if ((unsigned)yc >= (unsigned)WIMG) continue;
      #pragma unroll
      for (int dx = 0; dx < 2; ++dx) {
        int xc = 2 * px - 1 + dx;
        if ((unsigned)xc >= (unsigned)WIMG) continue;
        float s = 0.f;
        const int rb = 2 * threadIdx.y + dy;
        const int cb = 2 * threadIdx.x + dx;
        #pragma unroll
        for (int ci = 0; ci < 2; ++ci) {
          const float* srow = &sIn[ci * 1296];
          #pragma unroll
          for (int ky = 0; ky < 5; ++ky)
            #pragma unroll
            for (int kx = 0; kx < 5; ++kx)
              s += sW[ci * 25 + ky * 5 + kx] * srow[(rb + ky) * 36 + (cb + kx)];
        }
        if (s >= 5.0f) spike = 1.0f;
      }
    }
  }
  pooled1[((t * 4 + oc) * P1PAD + pyp) * P1PAD + pxp] = (half_t)spike;
}

// ---------------------------------------------------------------------------
// Kernel 2: conv2 as implicit-GEMM WMMA. K = 4ch*15ky*15kx = 900, packed two
// (c,ky) rows (15 taps + 1 zero slot) per K=32 chunk -> 30 WMMAs per tile.
// All pad/oc masking lives in the zero-padded B tile; A reads unconditional.
// Wave tile: M=16 output x, N=16 (8 real oc), one output row per wave.
// Emits spk2 = (pot >= 50) as u8. grid: (32, 63, 15), block 256.
// ---------------------------------------------------------------------------
__global__ __launch_bounds__(256) void conv2_wmma(
    const half_t* __restrict__ pooled1, const float* __restrict__ w2,
    unsigned char* __restrict__ spk2)
{
  // B tile: [n:16][pair:60][slot:16], slot 15 == 0, rows n>=8 == 0.
  __shared__ __align__(32) half_t sW2B[16 * 60 * 16];
  // A source: [c:4][y:22][x:30] + 8-element zero tail (pad-slot over-reads).
  __shared__ __align__(32) half_t sIn[4 * 22 * 30 + 8];
  const int t  = blockIdx.z;
  const int x0 = 16 * blockIdx.x;        // output col base == padded col base
  const int y0 = 8 * blockIdx.y;         // output row base == padded row base
  const int tid = threadIdx.x;

  for (int idx = tid; idx < 16 * 60 * 16; idx += 256) {
    int row = idx >> 4, slot = idx & 15;
    int n = row / 60, p = row - n * 60;
    half_t v = (half_t)0;
    if (n < 8 && slot < 15) v = (half_t)w2[n * 900 + p * 15 + slot];
    sW2B[idx] = v;
  }
  if (tid < 8) sIn[4 * 22 * 30 + tid] = (half_t)0;

#ifdef USE_TDM
  if (tid < 32) {
    const half_t* gsrc = pooled1 + (size_t)(t * 4) * P1PAD * P1PAD
                                 + (size_t)y0 * P1PAD + x0;
    tdm_load_tile_f16((uint32_t)(uintptr_t)(&sIn[0]), gsrc,
                      30, 22, 4, P1PAD, P1PAD, 4,
                      (uint64_t)P1PAD, (uint64_t)P1PAD * P1PAD);
    __builtin_amdgcn_s_wait_tensorcnt(0);
  }
#else
  for (int idx = tid; idx < 4 * 22 * 30; idx += 256) {
    int c = idx / 660, rem = idx - c * 660;
    int r = rem / 30, cc = rem - r * 30;
    int iy = y0 + r, ix = x0 + cc;
    half_t v = (half_t)0;
    if (iy < P1PAD && ix < P1PAD)
      v = pooled1[((t * 4 + c) * P1PAD + iy) * P1PAD + ix];
    sIn[idx] = v;
  }
#endif
  __syncthreads();

  const int wv    = tid >> 5;
  const int lane  = tid & 31;
  const int m     = lane & 15;                 // A-matrix M (output x offset)
  const int n     = lane & 15;                 // B-matrix N (out channel)
  const int kbase = (lane & 16) ? 8 : 0;       // per-lane K subset base
  const int yout  = y0 + wv;

  const int abase = wv * 30 + m + kbase;       // per-lane A base (elements)
  const int bbase = n * 960 + kbase;           // per-lane B base (elements)

  v8f acc = {};
  #pragma unroll
  for (int q = 0; q < 30; ++q) {
    const int p0 = 2 * q, p1 = 2 * q + 1;      // (c,ky) pair ids (constants)
    const int c0 = p0 / 15, ky0 = p0 - c0 * 15;
    const int c1 = p1 / 15, ky1 = p1 - c1 * 15;
    v8h a0 = load8u(&sIn[(c0 * 22 + ky0) * 30 + abase]);
    v8h a1 = load8u(&sIn[(c1 * 22 + ky1) * 30 + abase]);
    v8h b0 = *(const v8h*)&sW2B[bbase + p0 * 16];
    v8h b1 = *(const v8h*)&sW2B[bbase + p1 * 16];
    acc = wmma16x16x32(CAT16(a0, a1), CAT16(b0, b1), acc);
  }

  if (n < 8 && yout < C2OUT) {
    const int xb = x0 + ((lane & 16) ? 8 : 0); // D layout: M = r (+8 hi lanes)
    #pragma unroll
    for (int r = 0; r < 8; ++r) {
      const int x = xb + r;
      if (x < C2OUT)
        spk2[((size_t)(t * 8 + n) * C2OUT + yout) * C2OUT + x] =
            (acc[r] >= 50.0f) ? 1 : 0;
    }
  }
}

// ---------------------------------------------------------------------------
// Kernel 3: pool(spk2, 2x2 s2 pad1) -> padded 255x255 f16 (border zeroed).
// grid: (ceil(255*255/256), 120=t*8+c), block 256.
// ---------------------------------------------------------------------------
__global__ __launch_bounds__(256) void pool2_pad(
    const unsigned char* __restrict__ spk2, half_t* __restrict__ pooled2)
{
  const int idx = blockIdx.x * 256 + threadIdx.x;
  if (idx >= P2PAD * P2PAD) return;
  const int tc = blockIdx.y;
  const int pyp = idx / P2PAD, pxp = idx - pyp * P2PAD;
  float v = 0.f;
  if (pyp >= 2 && pyp < P2PAD - 2 && pxp >= 2 && pxp < P2PAD - 2) {
    const int py = pyp - 2, px = pxp - 2;
    #pragma unroll
    for (int dy = 0; dy < 2; ++dy) {
      int yc = 2 * py - 1 + dy;
      if ((unsigned)yc >= (unsigned)C2OUT) continue;
      #pragma unroll
      for (int dx = 0; dx < 2; ++dx) {
        int xc = 2 * px - 1 + dx;
        if ((unsigned)xc >= (unsigned)C2OUT) continue;
        if (spk2[((size_t)tc * C2OUT + yc) * C2OUT + xc]) v = 1.f;
      }
    }
  }
  pooled2[(size_t)tc * P2PAD * P2PAD + idx] = (half_t)v;
}

// ---------------------------------------------------------------------------
// Kernel 4: conv3 implicit-GEMM WMMA. K = 8ch*7ky*7kx = 392, packed four
// (c,ky) rows (7 taps + 1 zero slot) per K=32 chunk -> 14 WMMAs per tile.
// Fused fire@40; writes spk3/pot3 straight to d_out. grid (16,32,15).
// ---------------------------------------------------------------------------
__global__ __launch_bounds__(256) void conv3_wmma(
    const half_t* __restrict__ pooled2, const float* __restrict__ w3,
    float* __restrict__ out_spk, float* __restrict__ out_pot)
{
  // B tile: [n:16][pair:56][slot:8], slot 7 == 0, rows n>=2 == 0.
  __shared__ __align__(32) half_t sW3B[16 * 56 * 8];
  // A source: [c:8][y:14][x:22] + 8-element zero tail.
  __shared__ __align__(32) half_t sIn[8 * 14 * 22 + 8];
  const int t  = blockIdx.z;
  const int x0 = 16 * blockIdx.x;
  const int y0 = 8 * blockIdx.y;
  const int tid = threadIdx.x;

  for (int idx = tid; idx < 16 * 56 * 8; idx += 256) {
    int row = idx >> 3, slot = idx & 7;
    int n = row / 56, p = row - n * 56;
    half_t v = (half_t)0;
    if (n < 2 && slot < 7) v = (half_t)w3[n * 392 + p * 7 + slot];
    sW3B[idx] = v;
  }
  if (tid < 8) sIn[8 * 14 * 22 + tid] = (half_t)0;

#ifdef USE_TDM
  if (tid < 32) {
    const half_t* gsrc = pooled2 + (size_t)(t * 8) * P2PAD * P2PAD
                                 + (size_t)y0 * P2PAD + x0;
    tdm_load_tile_f16((uint32_t)(uintptr_t)(&sIn[0]), gsrc,
                      22, 14, 8, P2PAD, P2PAD, 8,
                      (uint64_t)P2PAD, (uint64_t)P2PAD * P2PAD);
    __builtin_amdgcn_s_wait_tensorcnt(0);
  }
#else
  for (int idx = tid; idx < 8 * 14 * 22; idx += 256) {
    int c = idx / 308, rem = idx - c * 308;
    int r = rem / 22, cc = rem - r * 22;
    int iy = y0 + r, ix = x0 + cc;
    half_t v = (half_t)0;
    if (iy < P2PAD && ix < P2PAD)
      v = pooled2[((t * 8 + c) * P2PAD + iy) * P2PAD + ix];
    sIn[idx] = v;
  }
#endif
  __syncthreads();

  const int wv    = tid >> 5;
  const int lane  = tid & 31;
  const int m     = lane & 15;
  const int n     = lane & 15;
  const int sb    = (lane & 16) ? 1 : 0;       // sub-block select (K>>3 bit 0)
  const int yout  = y0 + wv;

  const int abase = wv * 22 + m;               // per-lane A base (elements)

  v8f acc = {};
  #pragma unroll
  for (int q = 0; q < 14; ++q) {
    const int pA = 4 * q + sb;                 // K sub-blocks sb and sb+2
    const int pB = pA + 2;
    const int cA = pA / 7, kyA = pA - cA * 7;  // note: sb is lane-dependent,
    const int cB = pB / 7, kyB = pB - cB * 7;  // these stay cheap VALU ops
    v8h a0 = load8u(&sIn[(cA * 14 + kyA) * 22 + abase]);
    v8h a1 = load8u(&sIn[(cB * 14 + kyB) * 22 + abase]);
    v8h b0 = *(const v8h*)&sW3B[(n * 56 + pA) * 8];
    v8h b1 = *(const v8h*)&sW3B[(n * 56 + pB) * 8];
    acc = wmma16x16x32(CAT16(a0, a1), CAT16(b0, b1), acc);
  }

  if (n < 2 && yout < C3OUT) {
    const int xb = x0 + ((lane & 16) ? 8 : 0);
    #pragma unroll
    for (int r = 0; r < 8; ++r) {
      const int x = xb + r;
      if (x < C3OUT) {
        float pot = acc[r];
        float spk = (pot >= 40.0f) ? 1.0f : 0.0f;
        size_t o = ((size_t)(t * 2 + n) * C3OUT + yout) * C3OUT + x;
        out_spk[o] = spk;
        out_pot[o] = pot * spk;
      }
    }
  }
}

// ---------------------------------------------------------------------------
// Kernel 5: per-(c,h,w) "total" map for k-winners:
// total = sum_t(trunc) + sum_t(spk) * (max_t(trunc) * (T - sum_t(spk)))
// ---------------------------------------------------------------------------
__global__ __launch_bounds__(256) void compute_total(
    const float* __restrict__ spk, const float* __restrict__ pot,
    float* __restrict__ total)
{
  const int idx = blockIdx.x * 256 + threadIdx.x;
  if (idx >= 2 * HW3) return;
  float sum_spk = 0.f, sum_trunc = 0.f, mx = 0.f;
  for (int t = 0; t < T_STEPS; ++t) {
    size_t o = (size_t)t * 2 * HW3 + idx;
    float s = spk[o], p = pot[o];
    float tr = s * p;
    sum_spk += s; sum_trunc += tr; mx = fmaxf(mx, tr);
  }
  total[idx] = sum_trunc + sum_spk * (mx * ((float)T_STEPS - sum_spk));
}

// ---------------------------------------------------------------------------
// Kernel 6: sequential k=2 argmax + radius-5 inhibition in winning channel.
// Single block, 1024 threads. Winners written as floats (c,y,x or -1).
// ---------------------------------------------------------------------------
__global__ __launch_bounds__(1024) void kwinners(
    float* __restrict__ total, float* __restrict__ winners)
{
  __shared__ float sv[1024];
  __shared__ int   si[1024];
  __shared__ int   swin[4];
  const int tid = threadIdx.x;
  const int NTOT = 2 * HW3;
  for (int it = 0; it < 2; ++it) {
    float bv = -1.f; int bi = 0x7fffffff;
    for (int i = tid; i < NTOT; i += 1024) {
      float v = total[i];
      if (v > bv || (v == bv && i < bi)) { bv = v; bi = i; }
    }
    sv[tid] = bv; si[tid] = bi;
    __syncthreads();
    for (int ofs = 512; ofs > 0; ofs >>= 1) {
      if (tid < ofs) {
        float ov = sv[tid + ofs]; int oi = si[tid + ofs];
        if (ov > sv[tid] || (ov == sv[tid] && oi < si[tid])) {
          sv[tid] = ov; si[tid] = oi;
        }
      }
      __syncthreads();
    }
    if (tid == 0) {
      float v = sv[0]; int i = si[0];
      int c = i / HW3, rem = i - c * HW3;
      int y = rem / C3OUT, x = rem - y * C3OUT;
      int valid = (v > 0.f) ? 1 : 0;
      winners[3 * it + 0] = valid ? (float)c : -1.f;
      winners[3 * it + 1] = valid ? (float)y : -1.f;
      winners[3 * it + 2] = valid ? (float)x : -1.f;
      swin[0] = c; swin[1] = y; swin[2] = x; swin[3] = valid;
    }
    __syncthreads();
    if (swin[3]) {
      const int cw = swin[0], yw = swin[1], xw = swin[2];
      for (int i = tid; i < NTOT; i += 1024) {
        int c = i / HW3, rem = i - c * HW3;
        int y = rem / C3OUT, x = rem - y * C3OUT;
        int dy = y - yw; if (dy < 0) dy = -dy;
        int dx = x - xw; if (dx < 0) dx = -dx;
        if (c == cw && dy <= 5 && dx <= 5) total[i] = 0.f;
      }
    }
    __syncthreads();
  }
}

// ---------------------------------------------------------------------------
// Host-side launcher
// ---------------------------------------------------------------------------
extern "C" void kernel_launch(void* const* d_in, const int* in_sizes, int n_in,
                              void* d_out, int out_size, void* d_ws, size_t ws_size,
                              hipStream_t stream) {
  (void)in_sizes; (void)n_in; (void)out_size; (void)ws_size;
  const float* in = (const float*)d_in[0];
  const float* w1 = (const float*)d_in[1];
  const float* w2 = (const float*)d_in[2];
  const float* w3 = (const float*)d_in[3];
  float* out = (float*)d_out;

  char* ws = (char*)d_ws;
  size_t off = 0;
  auto carve = [&](size_t bytes) -> void* {
    void* p = ws + off;
    off += (bytes + 255) & ~(size_t)255;
    return p;
  };
  half_t* pooled1 = (half_t*)carve((size_t)T_STEPS * 4 * P1PAD * P1PAD * sizeof(half_t));
  unsigned char* spk2 = (unsigned char*)carve((size_t)T_STEPS * 8 * C2OUT * C2OUT);
  half_t* pooled2 = (half_t*)carve((size_t)T_STEPS * 8 * P2PAD * P2PAD * sizeof(half_t));
  float* total = (float*)carve((size_t)2 * HW3 * sizeof(float));

  float* out_spk = out;
  float* out_pot = out + N3;
  float* out_win = out + 2 * (size_t)N3;

  conv1_fire_pool<<<dim3(33, 33, 60), dim3(16, 16), 0, stream>>>(in, w1, pooled1);
  conv2_wmma<<<dim3(32, 63, 15), dim3(256), 0, stream>>>(pooled1, w2, spk2);
  pool2_pad<<<dim3((P2PAD * P2PAD + 255) / 256, 120), dim3(256), 0, stream>>>(spk2, pooled2);
  conv3_wmma<<<dim3(16, 32, 15), dim3(256), 0, stream>>>(pooled2, w3, out_spk, out_pot);
  compute_total<<<dim3((2 * HW3 + 255) / 256), dim3(256), 0, stream>>>(out_spk, out_pot, total);
  kwinners<<<dim3(1), dim3(1024), 0, stream>>>(total, out_win);
}
